// multi_level_expert_3762391351795
// MI455X (gfx1250) — matrix operand order repaired and checked by reading, not compiled, not run on hardware
//
#include <hip/hip_runtime.h>
#include <hip/hip_bf16.h>
#include <math.h>

// ---------------------------------------------------------------------------
// Types for CDNA5 WMMA (gfx1250, wave32)
// ---------------------------------------------------------------------------
typedef __attribute__((ext_vector_type(16))) __bf16 v16bf;
typedef __attribute__((ext_vector_type(8)))  float  v8f;

union BFrag {
    v16bf v;
    unsigned short s[16];
    uint4 q[2];
};

__device__ __forceinline__ unsigned short f2bf(float f) {
    unsigned int u = __float_as_uint(f);
    unsigned int r = (u + 0x7FFFu + ((u >> 16) & 1u)) >> 16;
    return (unsigned short)r;
}
__device__ __forceinline__ float bf2f(unsigned short s) {
    return __uint_as_float(((unsigned int)s) << 16);
}

// ---------------------------------------------------------------------------
// Problem constants
// ---------------------------------------------------------------------------
#define BATCH   1024
#define H1DIM   10816           // 16*26*26
#define H2DIM   18432           // 32*24*24
#define PIX1    676             // 26*26
#define PIX2    576             // 24*24
#define K2      144             // 16*3*3  (conv2 GEMM K)
#define K2PAD   160             // padded to multiple of 32
#define FC1K    18432
#define FC1N    128
#define FC1_SPLIT 8
#define FC1_KCHUNK (FC1K / FC1_SPLIT)   // 2304 = 72 WMMA steps

// im2col offset for conv2 gather: k -> (ci,ky,kx) -> h1-image offset.
// Returns -1 for padded K (k >= 144). Evaluated at compile time after
// loop unrolling (k is a constant per unrolled fragment element).
__host__ __device__ constexpr int im2col_off(int k) {
    return (k < K2) ? ((k / 9) * PIX1 + ((k % 9) / 3) * 26 + ((k % 9) % 3))
                    : -1;
}

// ---------------------------------------------------------------------------
// Kernel: f32 -> bf16 convert (weights)
// ---------------------------------------------------------------------------
__global__ void cvt_f32_bf16_kernel(const float* __restrict__ in,
                                    unsigned short* __restrict__ out, int n) {
    int i = blockIdx.x * blockDim.x + threadIdx.x;
    if (i < n) out[i] = f2bf(in[i]);
}

// ---------------------------------------------------------------------------
// Kernel: gating (logits over 2 experts -> softmax). One 256-thread block/row.
// ---------------------------------------------------------------------------
template <bool BF16IN>
__global__ void gate_kernel(const void* __restrict__ act_,
                            const float* __restrict__ wg,
                            float* __restrict__ gates, int K) {
    const int b = blockIdx.x;
    const int tid = threadIdx.x;
    float s0 = 0.f, s1 = 0.f;
    if (BF16IN) {
        const unsigned short* act = (const unsigned short*)act_ + (size_t)b * K;
        for (int k = tid; k < K; k += 256) {
            float a = bf2f(act[k]);
            s0 += a * wg[k * 2 + 0];
            s1 += a * wg[k * 2 + 1];
        }
    } else {
        const float* act = (const float*)act_ + (size_t)b * K;
        for (int k = tid; k < K; k += 256) {
            float a = act[k];
            s0 += a * wg[k * 2 + 0];
            s1 += a * wg[k * 2 + 1];
        }
    }
    __shared__ float r0[256], r1[256];
    r0[tid] = s0; r1[tid] = s1;
    __syncthreads();
    for (int off = 128; off > 0; off >>= 1) {
        if (tid < off) { r0[tid] += r0[tid + off]; r1[tid] += r1[tid + off]; }
        __syncthreads();
    }
    if (tid == 0) {
        float l0 = r0[0], l1 = r1[0];
        float m  = fmaxf(l0, l1);
        float e0 = __expf(l0 - m), e1 = __expf(l1 - m);
        float inv = 1.f / (e0 + e1);
        gates[b * 2 + 0] = e0 * inv;
        gates[b * 2 + 1] = e1 * inv;
    }
}

// ---------------------------------------------------------------------------
// Kernel: conv1 (Cin=1, 3x3 VALID, 16ch) + ReLU + gated expert sum -> h1 bf16
// ---------------------------------------------------------------------------
__global__ void conv1_moe_kernel(const float* __restrict__ x,
                                 const float* __restrict__ w,   // [2,16,1,3,3]
                                 const float* __restrict__ bias,// [2,16]
                                 const float* __restrict__ g,   // gates1 [B,2]
                                 unsigned short* __restrict__ h1) {
    int t = blockIdx.x * 256 + threadIdx.x;
    const int TOT = BATCH * H1DIM;
    if (t >= TOT) return;
    int b   = t / H1DIM;
    int rem = t - b * H1DIM;
    int c   = rem / PIX1;
    int p   = rem - c * PIX1;
    int oy  = p / 26;
    int ox  = p - oy * 26;
    const float* xi = x + b * 784 + oy * 28 + ox;
    const float* w0 = w + c * 9;
    const float* w1 = w + (16 + c) * 9;
    float a0 = bias[c], a1 = bias[16 + c];
#pragma unroll
    for (int ky = 0; ky < 3; ++ky)
#pragma unroll
        for (int kx = 0; kx < 3; ++kx) {
            float pv = xi[ky * 28 + kx];
            a0 += pv * w0[ky * 3 + kx];
            a1 += pv * w1[ky * 3 + kx];
        }
    float v = g[b * 2 + 0] * fmaxf(a0, 0.f) + g[b * 2 + 1] * fmaxf(a1, 0.f);
    h1[t] = f2bf(v);
}

// ---------------------------------------------------------------------------
// Kernel: conv2 as implicit GEMM via WMMA bf16.
//   M = B*24*24 pixels, N = 32 channels, K = 144 (pad 160), both experts.
// One wave: 16-pixel M tile x 32 channels x 2 experts (4 accumulators).
// A-gather offsets are compile-time constants (selected per lane-half),
// so the inner loop is loads + cndmask + wmma, no div/mod chains.
// 576 % 16 == 0 -> all 16 rows of a tile share the same batch image.
// ---------------------------------------------------------------------------
__global__ void conv2_moe_wmma_kernel(const unsigned short* __restrict__ h1,
                                      const unsigned short* __restrict__ w2, // bf16 [2*32,144]
                                      const float* __restrict__ bias,        // [2,32]
                                      const float* __restrict__ g,           // gates2 [B,2]
                                      unsigned short* __restrict__ h2) {
    const int wave  = blockIdx.x * 8 + (threadIdx.x >> 5);   // M tile id
    const int lane  = threadIdx.x & 31;
    const int mbase = wave * 16;

    // whole tile lives in one batch image
    const int ob    = mbase / PIX2;
    const int obase = mbase - ob * PIX2;      // tile-local pixel base

    // per-lane A row (pixel) within the image
    const int arem = obase + (lane & 15);
    const int aoy  = arem / 24;
    const int aox  = arem - aoy * 24;
    const unsigned short* pimg = h1 + (size_t)ob * H1DIM + aoy * 26 + aox;

    const bool hi    = (lane >= 16);
    const int  kbloc = hi ? 16 : 0;           // B K base within step
    const int  n_lo  = lane & 15;

    v8f acc00 = {}, acc01 = {}, acc10 = {}, acc11 = {};

#pragma unroll
    for (int ks = 0; ks < K2PAD; ks += 32) {
        // ---- A fragment: implicit-im2col gather, compile-time offsets ----
        BFrag a;
#pragma unroll
        for (int j = 0; j < 16; ++j) {
            const int klo = ks + j;                              // lanes 0-15
            const int khi = ks + 16 + j;                         // lanes 16-31
            const int oL = im2col_off(klo);
            const int oH = im2col_off(khi);
            unsigned short v = 0;
            int o = hi ? oH : oL;             // select between two constants
            if (o >= 0) {
                v = pimg[o];
            }
            a.s[j] = v;
        }
        // ---- B fragments: 2 experts x 2 N-tiles, contiguous-K loads ----
        const int kb = ks + kbloc;            // multiple of 16
        BFrag be[2][2];
#pragma unroll
        for (int e = 0; e < 2; ++e)
#pragma unroll
            for (int nt = 0; nt < 2; ++nt) {
                int n = nt * 16 + n_lo;
                if (kb < K2) {
                    const unsigned short* wp = w2 + (e * 32 + n) * K2 + kb;
                    be[e][nt].q[0] = *(const uint4*)(wp);
                    be[e][nt].q[1] = *(const uint4*)(wp + 8);
                } else {
                    be[e][nt].q[0] = make_uint4(0, 0, 0, 0);
                    be[e][nt].q[1] = make_uint4(0, 0, 0, 0);
                }
            }
        acc00 = __builtin_amdgcn_wmma_f32_16x16x32_bf16(false, a.v, false, be[0][0].v, (short)0, acc00, false, false);
        acc01 = __builtin_amdgcn_wmma_f32_16x16x32_bf16(false, a.v, false, be[0][1].v, (short)0, acc01, false, false);
        acc10 = __builtin_amdgcn_wmma_f32_16x16x32_bf16(false, a.v, false, be[1][0].v, (short)0, acc10, false, false);
        acc11 = __builtin_amdgcn_wmma_f32_16x16x32_bf16(false, a.v, false, be[1][1].v, (short)0, acc11, false, false);
    }

    // ---- epilogue: bias + ReLU + gate-weighted expert combine ----
    const float g0 = g[ob * 2 + 0];
    const float g1 = g[ob * 2 + 1];
    const int   mhi = hi ? 8 : 0;
    unsigned short* orow = h2 + (size_t)ob * H2DIM + obase + mhi;
    {
        const int n = n_lo;                   // channels 0..15
        const float b0 = bias[n], b1 = bias[32 + n];
        unsigned short* op = orow + n * PIX2;
#pragma unroll
        for (int r = 0; r < 8; ++r) {
            float v0 = fmaxf(acc00[r] + b0, 0.f);
            float v1 = fmaxf(acc10[r] + b1, 0.f);
            op[r] = f2bf(g0 * v0 + g1 * v1);
        }
    }
    {
        const int n = 16 + n_lo;              // channels 16..31
        const float b0 = bias[n], b1 = bias[32 + n];
        unsigned short* op = orow + n * PIX2;
#pragma unroll
        for (int r = 0; r < 8; ++r) {
            float v0 = fmaxf(acc01[r] + b0, 0.f);
            float v1 = fmaxf(acc11[r] + b1, 0.f);
            op[r] = f2bf(g0 * v0 + g1 * v1);
        }
    }
}

// ---------------------------------------------------------------------------
// Kernel: fc1 GEMM via WMMA bf16, 8-way K-split for occupancy.
// [1024,18432] x [18432,128]; each wave: one 16x16 tile x 2304-K chunk.
// Partials (f32) -> fc1_part[split][M][N]; reduced in fc1_reduce_kernel.
// ---------------------------------------------------------------------------
__global__ void fc1_wmma_kernel(const unsigned short* __restrict__ h2,   // bf16 [B,18432]
                                const unsigned short* __restrict__ w1,   // bf16 [128,18432]
                                float* __restrict__ part) {              // [8,B,128]
    const int wave  = blockIdx.x * 8 + (threadIdx.x >> 5);
    const int split = wave >> 9;          // 0..7
    const int rem   = wave & 511;
    const int mtile = rem >> 3;           // 0..63
    const int ntile = rem & 7;            // 0..7
    const int lane  = threadIdx.x & 31;

    const int arow = mtile * 16 + (lane & 15);
    const int bn   = ntile * 16 + (lane & 15);
    const bool hi  = (lane >= 16);
    const int albase = hi ? 16 : 0;
    const int ahbase = hi ? 24 : 8;
    const int kbloc  = hi ? 16 : 0;
    const int k0     = split * FC1_KCHUNK;

    const unsigned short* ap = h2 + (size_t)arow * FC1K + k0;
    const unsigned short* bp = w1 + (size_t)bn * FC1K + k0;

    v8f acc = {};
    for (int ks = 0; ks < FC1_KCHUNK; ks += 32) {
        BFrag a, b;
        a.q[0] = *(const uint4*)(ap + ks + albase);
        a.q[1] = *(const uint4*)(ap + ks + ahbase);
        b.q[0] = *(const uint4*)(bp + ks + kbloc);
        b.q[1] = *(const uint4*)(bp + ks + kbloc + 8);
        acc = __builtin_amdgcn_wmma_f32_16x16x32_bf16(false, a.v, false, b.v, (short)0, acc, false, false);
    }

    const int n   = ntile * 16 + (lane & 15);
    const int mhi = hi ? 8 : 0;
    float* op = part + (size_t)split * (BATCH * FC1N) + (size_t)(mtile * 16 + mhi) * FC1N + n;
#pragma unroll
    for (int r = 0; r < 8; ++r) op[r * FC1N] = acc[r];
}

__global__ void fc1_reduce_kernel(const float* __restrict__ part, // [8,B,128]
                                  const float* __restrict__ bias, // [128]
                                  float* __restrict__ out) {      // [B,128]
    int i = blockIdx.x * 256 + threadIdx.x;
    if (i >= BATCH * FC1N) return;
    float s = 0.f;
#pragma unroll
    for (int sp = 0; sp < FC1_SPLIT; ++sp) s += part[sp * (BATCH * FC1N) + i];
    out[i] = fmaxf(s + bias[i & (FC1N - 1)], 0.f);
}

// ---------------------------------------------------------------------------
// Kernel: fc2 + log_softmax. One thread per batch row (K=128, N=10).
// ---------------------------------------------------------------------------
__global__ void fc2_logsoftmax_kernel(const float* __restrict__ h,   // [B,128]
                                      const float* __restrict__ w,   // [10,128]
                                      const float* __restrict__ bias,// [10]
                                      float* __restrict__ out) {     // [B,10]
    int b = blockIdx.x * blockDim.x + threadIdx.x;
    if (b >= BATCH) return;
    const float* hp = h + (size_t)b * FC1N;
    float logit[10];
    float m = -INFINITY;
#pragma unroll
    for (int j = 0; j < 10; ++j) {
        float s = bias[j];
        const float* wp = w + j * FC1N;
        for (int k = 0; k < FC1N; ++k) s += hp[k] * wp[k];
        logit[j] = s;
        m = fmaxf(m, s);
    }
    float sum = 0.f;
#pragma unroll
    for (int j = 0; j < 10; ++j) sum += __expf(logit[j] - m);
    float lse = m + __logf(sum);
#pragma unroll
    for (int j = 0; j < 10; ++j) out[b * 10 + j] = logit[j] - lse;
}

// ---------------------------------------------------------------------------
// Launch
// ---------------------------------------------------------------------------
extern "C" void kernel_launch(void* const* d_in, const int* in_sizes, int n_in,
                              void* d_out, int out_size, void* d_ws, size_t ws_size,
                              hipStream_t stream) {
    const float* x       = (const float*)d_in[0];   // [1024,784]
    const float* w_gate1 = (const float*)d_in[1];   // [784,2]
    const float* conv1_w = (const float*)d_in[2];   // [2,16,1,3,3]
    const float* conv1_b = (const float*)d_in[3];   // [2,16]
    const float* w_gate2 = (const float*)d_in[4];   // [10816,2]
    const float* conv2_w = (const float*)d_in[5];   // [2,32,16,3,3]
    const float* conv2_b = (const float*)d_in[6];   // [2,32]
    const float* fc1_w   = (const float*)d_in[7];   // [128,18432]
    const float* fc1_b   = (const float*)d_in[8];   // [128]
    const float* fc2_w   = (const float*)d_in[9];   // [10,128]
    const float* fc2_b   = (const float*)d_in[10];  // [10]
    float* out = (float*)d_out;

    // Workspace layout (bytes), all offsets 256B aligned.  Total ~66.2 MB.
    char* ws = (char*)d_ws;
    float*          g1      = (float*)(ws + 0);                       //    8192 B
    float*          g2      = (float*)(ws + 8192);                    //    8192 B
    unsigned short* h1      = (unsigned short*)(ws + 16384);          // 22151168 B
    unsigned short* h2      = (unsigned short*)(ws + 22167552ull);    // 37748736 B
    float*          fc1_out = (float*)(ws + 59916288ull);             //   524288 B
    float*          fc1_par = (float*)(ws + 60440576ull);             //  4194304 B
    unsigned short* w2bf    = (unsigned short*)(ws + 64634880ull);    //    18432 B
    unsigned short* w1bf    = (unsigned short*)(ws + 64653312ull);    //  4718592 B

    // 0) convert weights to bf16
    {
        int n = 2 * 32 * 16 * 9;  // 9216
        cvt_f32_bf16_kernel<<<(n + 255) / 256, 256, 0, stream>>>(conv2_w, w2bf, n);
        int m = FC1N * FC1K;      // 2359296
        cvt_f32_bf16_kernel<<<(m + 255) / 256, 256, 0, stream>>>(fc1_w, w1bf, m);
    }
    // 1) gate1
    gate_kernel<false><<<BATCH, 256, 0, stream>>>((const void*)x, w_gate1, g1, 784);
    // 2) conv1 + relu + gated sum -> h1 (bf16)
    {
        int tot = BATCH * H1DIM;
        conv1_moe_kernel<<<(tot + 255) / 256, 256, 0, stream>>>(x, conv1_w, conv1_b, g1, h1);
    }
    // 3) gate2
    gate_kernel<true><<<BATCH, 256, 0, stream>>>((const void*)h1, w_gate2, g2, H1DIM);
    // 4) conv2 implicit-GEMM WMMA -> h2 (bf16)
    {
        int mtiles = (BATCH * PIX2) / 16;   // 36864 waves
        conv2_moe_wmma_kernel<<<mtiles / 8, 256, 0, stream>>>(h1, w2bf, conv2_b, g2, h2);
    }
    // 5) fc1 WMMA (K-split 8) -> partials -> reduce + bias + relu
    {
        int waves = (BATCH / 16) * (FC1N / 16) * FC1_SPLIT;  // 4096 waves
        fc1_wmma_kernel<<<waves / 8, 256, 0, stream>>>(h2, w1bf, fc1_par);
        int n = BATCH * FC1N;
        fc1_reduce_kernel<<<(n + 255) / 256, 256, 0, stream>>>(fc1_par, fc1_b, fc1_out);
    }
    // 6) fc2 + log_softmax -> out
    fc2_logsoftmax_kernel<<<BATCH / 256, 256, 0, stream>>>(fc1_out, fc2_w, fc2_b, out);
}
